// OptimAttn_58437325029955
// MI455X (gfx1250) — matrix operand order repaired and checked by reading, not compiled, not run on hardware
//
#include <hip/hip_runtime.h>

// MI455X (gfx1250, wave32) implementation.
// Compute-bound workload (~38 GFLOP, ~100MB traffic -> L2 resident at 192MB).
// All matmuls via v_wmma_f32_16x16x32_bf16; ADMM loop is LDS-resident with the
// Minv operand DMA'd into LDS by the Tensor Data Mover (TENSORcnt-tracked).

typedef __bf16 bf16;
typedef __attribute__((ext_vector_type(16))) bf16  v16bf;
typedef __attribute__((ext_vector_type(8)))  bf16  v8bf;
typedef __attribute__((ext_vector_type(2)))  bf16  v2bf;
typedef __attribute__((ext_vector_type(8)))  float v8f;
typedef unsigned int u32;
typedef __attribute__((ext_vector_type(4))) u32 u32x4;
typedef __attribute__((ext_vector_type(8))) int i32x8;
typedef __attribute__((ext_vector_type(4))) int i32x4;

#if defined(__has_builtin)
#if __has_builtin(__builtin_amdgcn_tensor_load_to_lds) && \
    __has_builtin(__builtin_amdgcn_s_wait_tensorcnt)
#define HAVE_TDM 1
#endif
#endif
#ifndef HAVE_TDM
#define HAVE_TDM 0
#endif

#define B_    32
#define N_    196
#define PD_   768
#define D_    512
#define NC_   1000
#define MPAD  224      // 196 padded to 7*32 for WMMA K and 14*16 for tiles
#define ROWS_ 16       // query rows per ADMM workgroup
#define NRB   13       // 13*16 = 208 >= 196 row-blocks per batch

// ---------------- block reductions (wave32) ----------------
__device__ __forceinline__ float block_sum(float x, float* red) {
#pragma unroll
  for (int o = 16; o > 0; o >>= 1) x += __shfl_xor(x, o, 32);
  const int wv = threadIdx.x >> 5, ln = threadIdx.x & 31;
  const int nw = blockDim.x >> 5;
  if (ln == 0) red[wv] = x;
  __syncthreads();
  float t = 0.f;
  for (int i = 0; i < nw; ++i) t += red[i];
  __syncthreads();
  return t;
}

__device__ __forceinline__ float block_max(float x, float* red) {
#pragma unroll
  for (int o = 16; o > 0; o >>= 1) x = fmaxf(x, __shfl_xor(x, o, 32));
  const int wv = threadIdx.x >> 5, ln = threadIdx.x & 31;
  const int nw = blockDim.x >> 5;
  if (ln == 0) red[wv] = x;
  __syncthreads();
  float t = -1e30f;
  for (int i = 0; i < nw; ++i) t = fmaxf(t, red[i]);
  __syncthreads();
  return t;
}

// ---------------- WMMA fragment helpers ----------------
// A 16x32 bf16: lane row=l16 holds 16 consecutive K (offset half*16),
// matching B. rowptr already includes row*ld + half*16.
__device__ __forceinline__ v16bf load_a_frag(const bf16* rowptr, int kb) {
  return *(const v16bf*)(rowptr + kb);
}
// B 32x16 bf16: lane holds column l16, 16 consecutive K (offset half*16) ->
// one contiguous 32B read from the Bt-source row. colptr includes col*ld + half*16.
__device__ __forceinline__ v16bf load_b_frag(const bf16* colptr, int kb) {
  return *(const v16bf*)(colptr + kb);
}
__device__ __forceinline__ v8f wmma_bf16(v16bf a, v16bf b, v8f c) {
  return __builtin_amdgcn_wmma_f32_16x16x32_bf16(false, a, false, b, (short)0, c,
                                                 false, false);
}

// ---------------- 0: weight convert + transpose fp32->bf16 ----------------
__global__ void wconv_k(const float* __restrict__ Wq, const float* __restrict__ Wv,
                        bf16* __restrict__ Wqt, bf16* __restrict__ Wvt) {
  const int idx = blockIdx.x * 256 + threadIdx.x;     // 0 .. 768*512-1
  const float* W = blockIdx.y ? Wv : Wq;
  bf16* Wt = blockIdx.y ? Wvt : Wqt;
  const int k = idx / D_, n = idx - k * D_;
  Wt[(size_t)n * PD_ + k] = (bf16)W[(size_t)k * D_ + n];
}

// ---------------- 1: patchify + LayerNorm + pos_emb -> bf16 X ----------------
__global__ void patch_ln_k(const float* __restrict__ img, const float* __restrict__ g,
                           const float* __restrict__ bta, const float* __restrict__ pos,
                           bf16* __restrict__ Xb) {
  __shared__ float red[8];
  const int n = blockIdx.x, b = blockIdx.y, tid = threadIdx.x;
  const int hh = n / 14, ww = n - hh * 14;
  float v[3];
  float s = 0.f;
#pragma unroll
  for (int e = 0; e < 3; ++e) {
    const int k = tid + e * 256;
    const int c = k >> 8, ph = (k >> 4) & 15, pw = k & 15;
    float x = img[(((size_t)b * 3 + c) * 224 + (hh * 16 + ph)) * 224 + (ww * 16 + pw)];
    v[e] = x; s += x;
  }
  const float mu = block_sum(s, red) * (1.0f / 768.0f);
  float s2 = 0.f;
#pragma unroll
  for (int e = 0; e < 3; ++e) { float d = v[e] - mu; s2 += d * d; }
  const float var = block_sum(s2, red) * (1.0f / 768.0f);
  const float rstd = rsqrtf(var + 1e-5f);
#pragma unroll
  for (int e = 0; e < 3; ++e) {
    const int k = tid + e * 256;
    float y = (v[e] - mu) * rstd * g[k] + bta[k] + pos[(size_t)n * 768 + k];
    Xb[((size_t)b * N_ + n) * PD_ + k] = (bf16)y;
  }
}

// ---------------- 2: Q/V projection GEMM [6272,768]x[768,512] ----------------
__global__ void __launch_bounds__(256)
qv_gemm(const bf16* __restrict__ Xb, const bf16* __restrict__ Wqt,
        const bf16* __restrict__ Wvt, const float* __restrict__ bq,
        const float* __restrict__ bv, float* __restrict__ Yq,
        float* __restrict__ Yv) {
  const int lane = threadIdx.x & 31, wave = threadIdx.x >> 5;
  const int half = lane >> 4, l16 = lane & 15;
  const int rt = blockIdx.x;                 // 392 row tiles (exact: 6272/16)
  const int dt = blockIdx.y * 8 + wave;      // 32 col tiles
  const int qv = blockIdx.z;
  const bf16* Wt = qv ? Wvt : Wqt;
  const float* bias = qv ? bv : bq;
  float* Y = qv ? Yv : Yq;
  const bf16* arow = Xb + (size_t)(rt * 16 + l16) * PD_ + half * 16;
  const bf16* brow = Wt + (size_t)(dt * 16 + l16) * PD_ + half * 16;
  v8f acc = {0.f, 0.f, 0.f, 0.f, 0.f, 0.f, 0.f, 0.f};
  for (int kb = 0; kb < PD_; kb += 32) {
    __builtin_prefetch(arow + kb + 64, 0, 1);   // global_prefetch_b8
    __builtin_prefetch(brow + kb + 64, 0, 1);
    acc = wmma_bf16(load_a_frag(arow, kb), load_b_frag(brow, kb), acc);
  }
  const int col = dt * 16 + l16;
  const float bb = bias[col];
#pragma unroll
  for (int j = 0; j < 8; ++j) {
    const int r = rt * 16 + j + half * 8;
    Y[(size_t)r * D_ + col] = acc[j] + bb;
  }
}

// ---------------- 3: row LayerNorm for Q and V (V scaled by 1/196) -------------
__global__ void qv_ln_k(const float* __restrict__ Yq, const float* __restrict__ Yv,
                        const float* __restrict__ gq, const float* __restrict__ bq2,
                        const float* __restrict__ gv, const float* __restrict__ bv2,
                        bf16* __restrict__ Qb, bf16* __restrict__ Vb,
                        float* __restrict__ Vf) {
  __shared__ float red[8];
  const int row = blockIdx.x, which = blockIdx.y, tid = threadIdx.x;
  const float* Y = which ? Yv : Yq;
  const float x0 = Y[(size_t)row * D_ + tid];
  const float x1 = Y[(size_t)row * D_ + tid + 256];
  const float mu = block_sum(x0 + x1, red) * (1.0f / 512.0f);
  const float d0 = x0 - mu, d1 = x1 - mu;
  const float var = block_sum(d0 * d0 + d1 * d1, red) * (1.0f / 512.0f);
  const float rstd = rsqrtf(var + 1e-5f);
  if (which == 0) {
    Qb[(size_t)row * D_ + tid]       = (bf16)(d0 * rstd * gq[tid] + bq2[tid]);
    Qb[(size_t)row * D_ + tid + 256] = (bf16)(d1 * rstd * gq[tid + 256] + bq2[tid + 256]);
  } else {
    const float v0 = (d0 * rstd * gv[tid] + bv2[tid]) * (1.0f / 196.0f);
    const float v1 = (d1 * rstd * gv[tid + 256] + bv2[tid + 256]) * (1.0f / 196.0f);
    Vb[(size_t)row * D_ + tid] = (bf16)v0;  Vb[(size_t)row * D_ + tid + 256] = (bf16)v1;
    Vf[(size_t)row * D_ + tid] = v0;        Vf[(size_t)row * D_ + tid + 256] = v1;
  }
}

// ---------------- 4: Q1 = 2 V V^T  and  p = -2 Q V^T + lambda/m ----------------
__global__ void __launch_bounds__(64)
q1p_gemm(const bf16* __restrict__ Qb, const bf16* __restrict__ Vb,
         float* __restrict__ Q1, float* __restrict__ P) {
  const int lane = threadIdx.x & 31, wave = threadIdx.x >> 5;   // block = 64
  const int half = lane >> 4, l16 = lane & 15;
  const int rt = blockIdx.x, ct = blockIdx.y, b = blockIdx.z;
  int rg = rt * 16 + l16; if (rg > 195) rg = 195;
  int cg = ct * 16 + l16; if (cg > 195) cg = 195;
  const bf16* Abase = (wave ? Qb : Vb) + ((size_t)b * N_ + rg) * D_ + half * 16;
  const bf16* Bbase = Vb + ((size_t)b * N_ + cg) * D_ + half * 16;
  v8f acc = {0.f, 0.f, 0.f, 0.f, 0.f, 0.f, 0.f, 0.f};
  for (int kb = 0; kb < D_; kb += 32) {
    __builtin_prefetch(Abase + kb + 64, 0, 1);
    __builtin_prefetch(Bbase + kb + 64, 0, 1);
    acc = wmma_bf16(load_a_frag(Abase, kb), load_b_frag(Bbase, kb), acc);
  }
  const int col = ct * 16 + l16;
#pragma unroll
  for (int j = 0; j < 8; ++j) {
    const int r = rt * 16 + j + half * 8;
    if (r < 196 && col < 196) {
      const size_t o = (size_t)b * 38416 + (size_t)r * 196 + col;
      if (wave) P[o]  = -2.0f * acc[j] + (1.0f / 196.0f);
      else      Q1[o] =  2.0f * acc[j];
    }
  }
}

// ---------------- 5: Gauss-Jordan inverse of (Q1 + I), LDS-resident ------------
__global__ void __launch_bounds__(256)
ginv_k(const float* __restrict__ Q1, float* __restrict__ Minv) {
  extern __shared__ char dsm[];
  float* A  = (float*)dsm;          // [196][200]
  float* Iv = A + 196 * 200;        // [196][200]
  float* spinv = Iv + 196 * 200;
  const int b = blockIdx.x, tid = threadIdx.x;
  const float* src = Q1 + (size_t)b * 38416;
  for (int idx = tid; idx < 196 * 200; idx += 256) {
    const int r = idx / 200, c = idx - r * 200;
    float a = 0.f, iv = 0.f;
    if (c < 196) { a = src[r * 196 + c]; if (c == r) { a += 1.0f; iv = 1.0f; } }
    A[idx] = a; Iv[idx] = iv;
  }
  __syncthreads();
  const int wave = tid >> 5, lane = tid & 31;
  for (int pv = 0; pv < 196; ++pv) {
    if (tid == 0) *spinv = 1.0f / A[pv * 200 + pv];
    __syncthreads();
    const float pinv = *spinv;
    for (int c = tid; c < 196; c += 256) {
      A[pv * 200 + c]  *= pinv;
      Iv[pv * 200 + c] *= pinv;
    }
    __syncthreads();
    for (int r = wave; r < 196; r += 8) {
      if (r == pv) continue;
      const float f = A[r * 200 + pv];
      for (int c = lane; c < 196; c += 32) {
        A[r * 200 + c]  -= f * A[pv * 200 + c];
        Iv[r * 200 + c] -= f * Iv[pv * 200 + c];
      }
    }
    __syncthreads();
  }
  for (int idx = tid; idx < 38416; idx += 256) {
    const int r = idx / 196, c = idx - r * 196;
    Minv[(size_t)b * 38416 + idx] = Iv[r * 200 + c];
  }
}

// ---------------- 5b: Minv fp32 -> padded bf16 [224x224] (TDM source) ----------
__global__ void mconv_k(const float* __restrict__ Minv, bf16* __restrict__ Mib) {
  const int idx = blockIdx.x * 256 + threadIdx.x;    // 0..50175
  const int b = blockIdx.y;
  const int i = idx / MPAD, k = idx - i * MPAD;
  const float v = (i < 196 && k < 196) ? Minv[(size_t)b * 38416 + i * 196 + k] : 0.0f;
  Mib[(size_t)b * MPAD * MPAD + idx] = (bf16)v;
}

// ---------------- 6: ADMM — LDS-resident 50-iter loop, WMMA matmul -------------
// One WG (8 waves) owns 16 query rows of one batch. Minv (bf16, 224x224, 98KB)
// is DMA'd into LDS by the Tensor Data Mover (wave 0 issues, TENSORcnt-tracked,
// overlapped with z/u/p init by the other waves); z/u/p/rhs tiles live in LDS.
__global__ void __launch_bounds__(256)
admm_k(const bf16* __restrict__ Mib, const float* __restrict__ P,
       float* __restrict__ Spart) {
  extern __shared__ char dsm[];
  bf16*  MinvL = (bf16*)dsm;                          // 224*224 bf16 = 100352B (offset 0!)
  float* pL = (float*)(dsm + MPAD * MPAD * 2);        // 16*224 f32
  float* zL = pL + ROWS_ * MPAD;
  float* uL = zL + ROWS_ * MPAD;
  bf16*  rhsL = (bf16*)(uL + ROWS_ * MPAD);           // 16*224 bf16
  const int rb = blockIdx.x, b = blockIdx.y;
  const int tid = threadIdx.x;
  const int lane = tid & 31, wave = tid >> 5;
  const int half = lane >> 4, l16 = lane & 15;
  const float* Pg = P + (size_t)b * 38416;
  const bf16* Mg = Mib + (size_t)b * MPAD * MPAD;

#if HAVE_TDM
  if (wave == 0) {
    // Tensor DMA descriptor (ISA 8.3/8.4): 2D 224x224 bf16 tile, global->LDS.
    const unsigned long long ga = (unsigned long long)(uintptr_t)Mg;
    u32x4 g0 = {1u,                                   // count=1, user mode
                0u,                                   // lds_addr = 0 (MinvL base)
                (u32)(ga & 0xFFFFFFFFu),              // global_addr[31:0]
                (u32)((ga >> 32) & 0x1FFFFFFu) | (2u << 30)}; // addr[56:32], type=2
    i32x8 g1 = {0x00010000,                           // data_size=1 (2 bytes/elem)
                (int)(224u << 16),                    // tensor_dim0 = 224 (bits 79:48)
                (int)(224u << 16),                    // tensor_dim1 = 224 (bits 111:80)
                (int)(224u << 16),                    // tile_dim0 = 224 (bits 127:112)
                224,                                  // tile_dim1 = 224 (bits 143:128)
                224,                                  // tensor_dim0_stride = 224
                0, 0};
    i32x4 gz = {0, 0, 0, 0};
#if __clang_major__ >= 23
    i32x8 gx = {0, 0, 0, 0, 0, 0, 0, 0};
    __builtin_amdgcn_tensor_load_to_lds(g0, g1, gz, gz, gx, 0);
#else
    __builtin_amdgcn_tensor_load_to_lds(g0, g1, gz, gz, 0);
#endif
  }
#else
  {
    const u32* srcw = (const u32*)Mg;
    u32* dstw = (u32*)MinvL;
    for (int i = tid; i < MPAD * MPAD / 2; i += 256) dstw[i] = srcw[i];
  }
#endif

  // z/u/p init (overlaps with the tensor DMA)
  for (int idx = tid; idx < ROWS_ * MPAD; idx += 256) {
    const int r = idx / MPAD, k = idx - r * MPAD;
    const int n = rb * ROWS_ + r;
    pL[idx] = (n < 196 && k < 196) ? Pg[n * 196 + k] : 0.0f;
    zL[idx] = 0.0f; uL[idx] = 0.0f;
  }
#if HAVE_TDM
  if (wave == 0) __builtin_amdgcn_s_wait_tensorcnt(0);
#endif
  __syncthreads();

  const bf16* arow = rhsL + l16 * MPAD + half * 16;
  for (int it = 0; it < 50; ++it) {
    // rhs = rho*(z-u) - p, rho = 1 (packed 2x bf16 -> ds_store_b32)
    for (int idx = tid; idx < ROWS_ * MPAD / 2; idx += 256) {
      const int f = idx * 2;
      v2bf pv;
      pv.x = (bf16)(zL[f]     - uL[f]     - pL[f]);
      pv.y = (bf16)(zL[f + 1] - uL[f + 1] - pL[f + 1]);
      *(v2bf*)(rhsL + f) = pv;
    }
    __syncthreads();
    // A fragments are shared by all i-tiles of this wave: hoist once per iter.
    v16bf afr[7];
#pragma unroll
    for (int i = 0; i < 7; ++i) afr[i] = load_a_frag(arow, 32 * i);
    for (int t = wave; t < 14; t += 8) {          // 14 i-tiles over 8 waves
      const bf16* brow = MinvL + (size_t)(t * 16 + l16) * MPAD + half * 16;
      v8f acc = {0.f, 0.f, 0.f, 0.f, 0.f, 0.f, 0.f, 0.f};
#pragma unroll
      for (int i = 0; i < 7; ++i)                 // x = rhs @ Minv^T
        acc = wmma_bf16(afr[i], load_b_frag(brow, 32 * i), acc);
      const int col = t * 16 + l16;
#pragma unroll
      for (int j = 0; j < 8; ++j) {               // z = clip(x+u); u += x - z
        const int r = j + half * 8;
        const float x = acc[j];
        const float u = uL[r * MPAD + col];
        const float zn = fminf(fmaxf(x + u, 0.0f), 1.0f);
        zL[r * MPAD + col] = zn;
        uL[r * MPAD + col] = u + x - zn;
      }
    }
    __syncthreads();
  }

  // coeffs = z / (sum|z|+1e-10); emit per-WG column sums (deterministic slots)
  if (tid < ROWS_) {
    float s = 0.f;
    for (int k = 0; k < MPAD; ++k) s += fabsf(zL[tid * MPAD + k]);
    pL[tid] = 1.0f / (s + 1e-10f);
  }
  __syncthreads();
  const int nvalid = (196 - rb * ROWS_) < ROWS_ ? (196 - rb * ROWS_) : ROWS_;
  for (int col = tid; col < 196; col += 256) {
    float a = 0.f;
    for (int r = 0; r < nvalid; ++r) a += zL[r * MPAD + col] * pL[r];
    Spart[((size_t)b * NRB + rb) * N_ + col] = a;
  }
}

// ---------------- 7: head — pooled = (mean coeffs) @ V, LN, FC, softmax --------
__global__ void __launch_bounds__(256)
head_k(const float* __restrict__ Spart, const float* __restrict__ Vf,
       const float* __restrict__ g, const float* __restrict__ bta,
       const float* __restrict__ W, const float* __restrict__ bb,
       float* __restrict__ out) {
  __shared__ float sv[196];
  __shared__ float lnv[512];
  __shared__ float logits[1000];
  __shared__ float red[8];
  const int b = blockIdx.x, tid = threadIdx.x;
  for (int m = tid; m < 196; m += 256) {
    float a = 0.f;
    for (int rb2 = 0; rb2 < NRB; ++rb2) a += Spart[((size_t)b * NRB + rb2) * N_ + m];
    sv[m] = a * (1.0f / 196.0f);
  }
  __syncthreads();
  float p0 = 0.f, p1 = 0.f;
  {
    const float* vp = Vf + (size_t)b * N_ * D_;
    for (int m = 0; m < 196; ++m) {
      const float w = sv[m];
      p0 += w * vp[m * D_ + tid];
      p1 += w * vp[m * D_ + tid + 256];
    }
  }
  const float mu = block_sum(p0 + p1, red) * (1.0f / 512.0f);
  const float d0 = p0 - mu, d1 = p1 - mu;
  const float var = block_sum(d0 * d0 + d1 * d1, red) * (1.0f / 512.0f);
  const float rstd = rsqrtf(var + 1e-5f);
  lnv[tid]       = d0 * rstd * g[tid] + bta[tid];
  lnv[tid + 256] = d1 * rstd * g[tid + 256] + bta[tid + 256];
  __syncthreads();
  for (int c = tid; c < NC_; c += 256) {
    float a = bb[c];
    for (int d = 0; d < D_; ++d) a += lnv[d] * W[(size_t)d * NC_ + c];
    logits[c] = a;
  }
  __syncthreads();
  float mx = -1e30f;
  for (int c = tid; c < NC_; c += 256) mx = fmaxf(mx, logits[c]);
  mx = block_max(mx, red);
  float se = 0.f;
  for (int c = tid; c < NC_; c += 256) se += __expf(logits[c] - mx);
  se = block_sum(se, red);
  const float inv = 1.0f / se;
  for (int c = tid; c < NC_; c += 256)
    out[(size_t)b * NC_ + c] = __expf(logits[c] - mx) * inv;
}

// ---------------- launch ----------------
extern "C" void kernel_launch(void* const* d_in, const int* in_sizes, int n_in,
                              void* d_out, int out_size, void* d_ws, size_t ws_size,
                              hipStream_t stream) {
  (void)in_sizes; (void)n_in; (void)out_size; (void)ws_size;
  const float* img   = (const float*)d_in[0];
  const float* lpg   = (const float*)d_in[1];
  const float* lpb   = (const float*)d_in[2];
  const float* wq_w  = (const float*)d_in[3];
  const float* wq_b  = (const float*)d_in[4];
  const float* lnq_g = (const float*)d_in[5];
  const float* lnq_b = (const float*)d_in[6];
  const float* wv_w  = (const float*)d_in[7];
  const float* wv_b  = (const float*)d_in[8];
  const float* lnv_g = (const float*)d_in[9];
  const float* lnv_b = (const float*)d_in[10];
  const float* pos   = (const float*)d_in[11];
  const float* mlg   = (const float*)d_in[12];
  const float* mlb   = (const float*)d_in[13];
  const float* mw    = (const float*)d_in[14];
  const float* mb    = (const float*)d_in[15];
  float* out = (float*)d_out;

  char* w = (char*)d_ws;
  size_t off = 0;
  auto take = [&](size_t bytes) -> char* {
    char* p = w + off; off += (bytes + 255) & ~(size_t)255; return p;
  };
  bf16*  Xb  = (bf16*) take((size_t)B_ * N_ * PD_ * 2);
  bf16*  Wqt = (bf16*) take((size_t)D_ * PD_ * 2);
  bf16*  Wvt = (bf16*) take((size_t)D_ * PD_ * 2);
  float* Yq  = (float*)take((size_t)B_ * N_ * D_ * 4);
  float* Yv  = (float*)take((size_t)B_ * N_ * D_ * 4);
  bf16*  Qb  = (bf16*) take((size_t)B_ * N_ * D_ * 2);
  bf16*  Vb  = (bf16*) take((size_t)B_ * N_ * D_ * 2);
  float* Vf  = (float*)take((size_t)B_ * N_ * D_ * 4);
  float* Q1  = (float*)take((size_t)B_ * N_ * N_ * 4);
  float* Pm  = (float*)take((size_t)B_ * N_ * N_ * 4);
  float* Mi  = (float*)take((size_t)B_ * N_ * N_ * 4);
  bf16*  Mib = (bf16*) take((size_t)B_ * MPAD * MPAD * 2);
  float* Sp  = (float*)take((size_t)B_ * NRB * N_ * 4);

  wconv_k   <<<dim3(1536, 2),    256, 0, stream>>>(wq_w, wv_w, Wqt, Wvt);
  patch_ln_k<<<dim3(196, 32),    256, 0, stream>>>(img, lpg, lpb, pos, Xb);
  qv_gemm   <<<dim3(392, 4, 2),  256, 0, stream>>>(Xb, Wqt, Wvt, wq_b, wv_b, Yq, Yv);
  qv_ln_k   <<<dim3(6272, 2),    256, 0, stream>>>(Yq, Yv, lnq_g, lnq_b, lnv_g, lnv_b,
                                                   Qb, Vb, Vf);
  q1p_gemm  <<<dim3(13, 13, 32),  64, 0, stream>>>(Qb, Vb, Q1, Pm);
  ginv_k    <<<dim3(32), 256, 2 * 196 * 200 * 4 + 16, stream>>>(Q1, Mi);
  mconv_k   <<<dim3(196, 32),    256, 0, stream>>>(Mi, Mib);
  admm_k    <<<dim3(13, 32), 256,
               MPAD * MPAD * 2 + 3 * ROWS_ * MPAD * 4 + ROWS_ * MPAD * 2,
               stream>>>(Mib, Pm, Sp);
  head_k    <<<dim3(32), 256, 0, stream>>>(Sp, Vf, mlg, mlb, mw, mb, out);
}
